// EfficientHistoryModel_15607911153903
// MI455X (gfx1250) — compile-verified, matrix operand order checked by reading, not run on hardware
//
#include <hip/hip_runtime.h>
#include <math.h>

#define B_  512
#define S_  100
#define V_  100000
#define D_  128
#define H_  8
#define HD_ 16
#define FF_ 256
#define NL_ 3

typedef __attribute__((ext_vector_type(16))) _Float16 v16h;
typedef __attribute__((ext_vector_type(8)))  float    v8f;

// ---------------------------------------------------------------------------
// 16x32 f32->f16 tile fetch in the CDNA5 WMMA 16-bit A layout:
//   lane L: row = row0 + (L&15)
//   element e: K = k0 + (e>=8 ? 16 : 0) + (L>=16 ? 8 : 0) + (e&7)
// => two contiguous 8-float runs per lane -> 4 x float4 loads.
// W is stored row-major (N x K), so B tiles use the identical fetch.
// ---------------------------------------------------------------------------
__device__ __forceinline__ v16h load_tile16x32(const float* __restrict__ base,
                                               int ld, int row0, int k0, int lane) {
  const float* p = base + (size_t)(row0 + (lane & 15)) * ld + k0 + ((lane >> 4) ? 8 : 0);
  const float4 f0 = *(const float4*)(p + 0);
  const float4 f1 = *(const float4*)(p + 4);
  const float4 f2 = *(const float4*)(p + 16);
  const float4 f3 = *(const float4*)(p + 20);
  v16h r;
  r[0]  = (_Float16)f0.x; r[1]  = (_Float16)f0.y; r[2]  = (_Float16)f0.z; r[3]  = (_Float16)f0.w;
  r[4]  = (_Float16)f1.x; r[5]  = (_Float16)f1.y; r[6]  = (_Float16)f1.z; r[7]  = (_Float16)f1.w;
  r[8]  = (_Float16)f2.x; r[9]  = (_Float16)f2.y; r[10] = (_Float16)f2.z; r[11] = (_Float16)f2.w;
  r[12] = (_Float16)f3.x; r[13] = (_Float16)f3.y; r[14] = (_Float16)f3.z; r[15] = (_Float16)f3.w;
  return r;
}

// ---------------------------------------------------------------------------
// C = A(MxK) @ W^T (W stored NxK row-major) + bias, optional exact GELU.
// Block = 256 threads (8 wave32). Block tile = 64x64:
//   wave w -> M subtile (w&3)*16, N subtile (w>>2)*32.
// Each wave computes 16x32 = two 16x16 WMMA accumulators sharing the A tile.
// K is a template constant (128 or 256) so the MAC loop fully unrolls:
// 2*(K/32) v_wmma per wave iteration chain, no loop SALU.
// Requires: M % 64 == 0, N % 64 == 0 (true for every GEMM here).
// ---------------------------------------------------------------------------
template <int GELU, int K>
__global__ void gemm_bias_kernel(const float* __restrict__ A, const float* __restrict__ W,
                                 const float* __restrict__ bias, float* __restrict__ C,
                                 int M, int N) {
  const int lane = threadIdx.x & 31;
  const int wave = threadIdx.x >> 5;
  const int m0 = blockIdx.x * 64 + (wave & 3) * 16;
  const int n0 = blockIdx.y * 64 + (wave >> 2) * 32;

  v8f acc0 = {};
  v8f acc1 = {};
#pragma unroll
  for (int k0 = 0; k0 < K; k0 += 32) {
    const v16h a  = load_tile16x32(A, K, m0, k0, lane);
    const v16h b0 = load_tile16x32(W, K, n0, k0, lane);
    const v16h b1 = load_tile16x32(W, K, n0 + 16, k0, lane);
    acc0 = __builtin_amdgcn_wmma_f32_16x16x32_f16(false, a, false, b0,
                                                  (short)0, acc0, false, false);
    acc1 = __builtin_amdgcn_wmma_f32_16x16x32_f16(false, a, false, b1,
                                                  (short)0, acc1, false, false);
  }

  // C/D layout: VGPR r -> row = m0 + r + 8*(lane>=16), col = n0 + (lane&15)
  const int col0  = n0 + (lane & 15);
  const int rbase = m0 + ((lane >> 4) ? 8 : 0);
  const float bb0 = bias[col0];
  const float bb1 = bias[col0 + 16];
#pragma unroll
  for (int r = 0; r < 8; ++r) {
    float v0 = acc0[r] + bb0;
    float v1 = acc1[r] + bb1;
    if (GELU) {
      v0 = 0.5f * v0 * (1.0f + erff(v0 * 0.70710678118654752f));
      v1 = 0.5f * v1 * (1.0f + erff(v1 * 0.70710678118654752f));
    }
    const size_t row = (size_t)(rbase + r) * N;
    C[row + col0]      = v0;
    C[row + col0 + 16] = v1;
  }
}

// ---------------------------------------------------------------------------
// Embedding gather + temporal features + LayerNorm + sinusoidal PE.
// One token per 128-thread block (thread = channel).
// ---------------------------------------------------------------------------
__global__ void embed_kernel(const float* __restrict__ loc_emb, const float* __restrict__ user_emb,
                             const float* __restrict__ temp_W, const float* __restrict__ temp_b,
                             const float* __restrict__ ln_s, const float* __restrict__ ln_b,
                             const float* __restrict__ start_min, const float* __restrict__ dur,
                             const float* __restrict__ diff, const int* __restrict__ loc_seq,
                             const int* __restrict__ user_seq, const int* __restrict__ weekday,
                             float* __restrict__ x) {
  const int tok = blockIdx.x;
  const int t   = tok % S_;
  const int d   = threadIdx.x;

  const float twopi = 6.283185307179586f;
  const float tr = start_min[tok] * (twopi / 1440.0f);
  const float wd = (float)weekday[tok] * (twopi / 7.0f);
  const float f0 = sinf(tr), f1 = cosf(tr), f2 = log1pf(dur[tok]) * 0.125f;
  const float f3 = sinf(wd), f4 = cosf(wd), f5 = diff[tok] * (1.0f / 7.0f);

  float val;
  if (d < 64) {
    val = loc_emb[(size_t)loc_seq[tok] * 64 + d];
  } else if (d < 80) {
    val = user_emb[(size_t)user_seq[tok] * 16 + (d - 64)];
  } else {
    const float* w = temp_W + (d - 80) * 6;
    val = w[0]*f0 + w[1]*f1 + w[2]*f2 + w[3]*f3 + w[4]*f4 + w[5]*f5 + temp_b[d - 80];
  }

  __shared__ float red[D_];
  red[d] = val; __syncthreads();
  for (int s = 64; s > 0; s >>= 1) { if (d < s) red[d] += red[d + s]; __syncthreads(); }
  const float mean = red[0] * (1.0f / D_);
  __syncthreads();
  const float c = val - mean;
  red[d] = c * c; __syncthreads();
  for (int s = 64; s > 0; s >>= 1) { if (d < s) red[d] += red[d + s]; __syncthreads(); }
  const float var = red[0] * (1.0f / D_);

  float y = c * rsqrtf(var + 1e-5f) * ln_s[d] + ln_b[d];

  const int i = d >> 1;
  const float freq = expf(-(float)(2 * i) * (9.210340371976184f / 128.0f)); // ln(1e4)/128
  const float ang  = (float)t * freq;
  y += (d & 1) ? cosf(ang) : sinf(ang);

  x[(size_t)tok * D_ + d] = y;
}

// ---------------------------------------------------------------------------
// x = LayerNorm(x + r) * s + b   (one token per 128-thread block)
// ---------------------------------------------------------------------------
__global__ void add_ln_kernel(float* __restrict__ x, const float* __restrict__ r_,
                              const float* __restrict__ s_, const float* __restrict__ b_) {
  const int tok = blockIdx.x;
  const int d   = threadIdx.x;
  const size_t idx = (size_t)tok * D_ + d;
  const float val = x[idx] + r_[idx];

  __shared__ float red[D_];
  red[d] = val; __syncthreads();
  for (int s = 64; s > 0; s >>= 1) { if (d < s) red[d] += red[d + s]; __syncthreads(); }
  const float mean = red[0] * (1.0f / D_);
  __syncthreads();
  const float c = val - mean;
  red[d] = c * c; __syncthreads();
  for (int s = 64; s > 0; s >>= 1) { if (d < s) red[d] += red[d + s]; __syncthreads(); }
  const float var = red[0] * (1.0f / D_);

  x[idx] = c * rsqrtf(var + 1e-5f) * s_[d] + b_[d];
}

// ---------------------------------------------------------------------------
// Attention: one (batch, head) per block. K/V staged in LDS, two-pass softmax.
// qkv rows: [q(128) | k(128) | v(128)], head h at offset h*16.
// Output o laid out (b, t, h*16+d) so the out-proj GEMM reads it directly.
// ---------------------------------------------------------------------------
__global__ void attn_kernel(const float* __restrict__ qkv, const int* __restrict__ mask,
                            float* __restrict__ o) {
  const int b = blockIdx.x / H_;
  const int h = blockIdx.x % H_;
  const int tid = threadIdx.x;

  __shared__ float Ks[S_][HD_];
  __shared__ float Vs[S_][HD_];
  __shared__ float mk[S_];

  for (int idx = tid; idx < S_ * HD_; idx += blockDim.x) {
    const int t = idx / HD_, d = idx % HD_;
    const size_t row = (size_t)(b * S_ + t) * (3 * D_);
    Ks[t][d] = qkv[row + D_     + h * HD_ + d];
    Vs[t][d] = qkv[row + 2 * D_ + h * HD_ + d];
  }
  for (int t = tid; t < S_; t += blockDim.x) mk[t] = mask[b * S_ + t] ? 1.0f : 0.0f;
  __syncthreads();

  for (int q = tid; q < S_; q += blockDim.x) {
    float qv[HD_];
    const size_t qrow = (size_t)(b * S_ + q) * (3 * D_) + h * HD_;
#pragma unroll
    for (int d = 0; d < HD_; ++d) qv[d] = qkv[qrow + d];

    float mx = -INFINITY;
    for (int k = 0; k < S_; ++k) {
      if (mk[k] == 0.0f) continue;
      float s = 0.0f;
#pragma unroll
      for (int d = 0; d < HD_; ++d) s += qv[d] * Ks[k][d];
      mx = fmaxf(mx, s * 0.25f);
    }

    float acc[HD_];
#pragma unroll
    for (int d = 0; d < HD_; ++d) acc[d] = 0.0f;
    float sum = 0.0f;
    for (int k = 0; k < S_; ++k) {
      if (mk[k] == 0.0f) continue;
      float s = 0.0f;
#pragma unroll
      for (int d = 0; d < HD_; ++d) s += qv[d] * Ks[k][d];
      const float p = expf(s * 0.25f - mx);
      sum += p;
#pragma unroll
      for (int d = 0; d < HD_; ++d) acc[d] += p * Vs[k][d];
    }
    const float inv = 1.0f / sum;
    const size_t orow = (size_t)(b * S_ + q) * D_ + h * HD_;
#pragma unroll
    for (int d = 0; d < HD_; ++d) o[orow + d] = acc[d] * inv;
  }
}

// ---------------------------------------------------------------------------
// last[b, :] = x[b, sum(mask[b]) - 1, :]
// ---------------------------------------------------------------------------
__global__ void gather_last_kernel(const float* __restrict__ x, const int* __restrict__ mask,
                                   float* __restrict__ last) {
  const int b = blockIdx.x;
  __shared__ int len;
  if (threadIdx.x == 0) {
    int l = 0;
    for (int t = 0; t < S_; ++t) l += (mask[b * S_ + t] != 0);
    len = l;
  }
  __syncthreads();
  last[(size_t)b * D_ + threadIdx.x] = x[((size_t)b * S_ + (len - 1)) * D_ + threadIdx.x];
}

// ---------------------------------------------------------------------------
// -inf fill of the (B, V) logits buffer, float4-vectorized (205 MB of HBM
// traffic -> keep it at streaming bandwidth). B*V % 4 == 0.
// ---------------------------------------------------------------------------
__global__ void fill_kernel(float4* __restrict__ out, long long n4) {
  long long i = (long long)blockIdx.x * blockDim.x + threadIdx.x;
  const long long stride = (long long)gridDim.x * blockDim.x;
  const float4 v = make_float4(-INFINITY, -INFINITY, -INFINITY, -INFINITY);
  for (; i < n4; i += stride) out[i] = v;
}

// ---------------------------------------------------------------------------
// First-occurrence candidate scoring: logits[b, loc] = <loc_emb[loc], query[b]>
// for the first masked occurrence of each loc in the sequence.
// ---------------------------------------------------------------------------
__global__ void score_kernel(const float* __restrict__ loc_emb, const int* __restrict__ loc_seq,
                             const int* __restrict__ mask, const float* __restrict__ query,
                             float* __restrict__ out) {
  const int b = blockIdx.x;
  for (int t = threadIdx.x; t < S_; t += blockDim.x) {
    if (!mask[b * S_ + t]) continue;
    const int loc = loc_seq[b * S_ + t];
    bool first = true;
    for (int t2 = 0; t2 < t; ++t2) {
      if (mask[b * S_ + t2] && loc_seq[b * S_ + t2] == loc) { first = false; break; }
    }
    if (!first) continue;
    const float* e  = loc_emb + (size_t)loc * 64;
    const float* qv = query + (size_t)b * 64;
    float s = 0.0f;
#pragma unroll 8
    for (int d = 0; d < 64; ++d) s += e[d] * qv[d];
    out[(size_t)b * V_ + loc] = s;
  }
}

// ---------------------------------------------------------------------------
extern "C" void kernel_launch(void* const* d_in, const int* in_sizes, int n_in,
                              void* d_out, int out_size, void* d_ws, size_t ws_size,
                              hipStream_t stream) {
  (void)in_sizes; (void)n_in; (void)out_size; (void)ws_size;

  const float* loc_emb_W  = (const float*)d_in[0];
  const float* user_emb_W = (const float*)d_in[1];
  const float* temp_W     = (const float*)d_in[2];
  const float* temp_b     = (const float*)d_in[3];
  const float* ln_in_s    = (const float*)d_in[4];
  const float* ln_in_b    = (const float*)d_in[5];
  const float* qkv_w      = (const float*)d_in[6];
  const float* qkv_b      = (const float*)d_in[7];
  const float* attn_out_w = (const float*)d_in[8];
  const float* attn_out_b = (const float*)d_in[9];
  const float* ln1_s      = (const float*)d_in[10];
  const float* ln1_b      = (const float*)d_in[11];
  const float* ff1_w      = (const float*)d_in[12];
  const float* ff1_b      = (const float*)d_in[13];
  const float* ff2_w      = (const float*)d_in[14];
  const float* ff2_b      = (const float*)d_in[15];
  const float* ln2_s      = (const float*)d_in[16];
  const float* ln2_b      = (const float*)d_in[17];
  const float* op1_w      = (const float*)d_in[18];
  const float* op1_b      = (const float*)d_in[19];
  const float* op2_w      = (const float*)d_in[20];
  const float* op2_b      = (const float*)d_in[21];
  const float* start_min  = (const float*)d_in[22];
  const float* dur        = (const float*)d_in[23];
  const float* diff       = (const float*)d_in[24];
  const int*   loc_seq    = (const int*)d_in[25];
  const int*   user_seq   = (const int*)d_in[26];
  const int*   weekday    = (const int*)d_in[27];
  const int*   mask       = (const int*)d_in[28];

  const int M = B_ * S_; // 51200 tokens

  // Workspace layout (floats). Total ~158 MB.
  float* x      = (float*)d_ws;
  float* qkvbuf = x      + (size_t)M * D_;        // 51200 x 384 (reused as 51200 x 256 FF mid)
  float* obuf   = qkvbuf + (size_t)M * 3 * D_;    // 51200 x 128 attention output
  float* pbuf   = obuf   + (size_t)M * D_;        // 51200 x 128 projection / FF out
  float* last   = pbuf   + (size_t)M * D_;        // 512 x 128
  float* mid    = last   + (size_t)B_ * D_;       // 512 x 256
  float* query  = mid    + (size_t)B_ * FF_;      // 512 x 64

  embed_kernel<<<M, D_, 0, stream>>>(loc_emb_W, user_emb_W, temp_W, temp_b,
                                     ln_in_s, ln_in_b, start_min, dur, diff,
                                     loc_seq, user_seq, weekday, x);

  for (int i = 0; i < NL_; ++i) {
    // QKV projection: 51200x128 @ (384x128)^T
    gemm_bias_kernel<0, 128><<<dim3(M / 64, (3 * D_) / 64), 256, 0, stream>>>(
        x, qkv_w + (size_t)i * 3 * D_ * D_, qkv_b + (size_t)i * 3 * D_, qkvbuf, M, 3 * D_);

    attn_kernel<<<B_ * H_, 128, 0, stream>>>(qkvbuf, mask, obuf);

    // Output projection: 51200x128 @ (128x128)^T
    gemm_bias_kernel<0, 128><<<dim3(M / 64, D_ / 64), 256, 0, stream>>>(
        obuf, attn_out_w + (size_t)i * D_ * D_, attn_out_b + (size_t)i * D_, pbuf, M, D_);

    add_ln_kernel<<<M, D_, 0, stream>>>(x, pbuf, ln1_s + i * D_, ln1_b + i * D_);

    // FF1 + GELU: 51200x128 @ (256x128)^T
    gemm_bias_kernel<1, 128><<<dim3(M / 64, FF_ / 64), 256, 0, stream>>>(
        x, ff1_w + (size_t)i * FF_ * D_, ff1_b + (size_t)i * FF_, qkvbuf, M, FF_);

    // FF2: 51200x256 @ (128x256)^T
    gemm_bias_kernel<0, 256><<<dim3(M / 64, D_ / 64), 256, 0, stream>>>(
        qkvbuf, ff2_w + (size_t)i * D_ * FF_, ff2_b + (size_t)i * D_, pbuf, M, D_);

    add_ln_kernel<<<M, D_, 0, stream>>>(x, pbuf, ln2_s + i * D_, ln2_b + i * D_);
  }

  gather_last_kernel<<<B_, D_, 0, stream>>>(x, mask, last);

  // op1 + GELU: 512x128 @ (256x128)^T ; op2: 512x256 @ (64x256)^T
  gemm_bias_kernel<1, 128><<<dim3(B_ / 64, FF_ / 64), 256, 0, stream>>>(last, op1_w, op1_b, mid, B_, FF_);
  gemm_bias_kernel<0, 256><<<dim3(B_ / 64, 64 / 64), 256, 0, stream>>>(mid, op2_w, op2_b, query, B_, 64);

  fill_kernel<<<8192, 256, 0, stream>>>((float4*)d_out, (long long)(B_ * (long long)V_) / 4);
  score_kernel<<<B_, 128, 0, stream>>>(loc_emb_W, loc_seq, mask, query, (float*)d_out);
}
